// QuantizedLinear_44427141710171
// MI455X (gfx1250) — 8-bit matrix operand order repaired and checked by reading, compiled, not run on hardware
//
#include <hip/hip_runtime.h>

typedef __attribute__((ext_vector_type(8))) int v8i;
typedef int v4i_b __attribute__((vector_size(16)));
typedef __attribute__((address_space(1))) v4i_b* gas_v4i_ptr;
typedef __attribute__((address_space(3))) v4i_b* las_v4i_ptr;

#define BM 128
#define BN 128
#define BK 64
#define LDA_S 72
#define LDB_S 80

#if defined(__has_builtin)
#if __has_builtin(__builtin_amdgcn_global_load_async_to_lds_b128) && \
    __has_builtin(__builtin_amdgcn_s_wait_asynccnt)
#define USE_ASYNC_LDS 1
#endif
#endif
#ifndef USE_ASYNC_LDS
#define USE_ASYNC_LDS 0
#endif

__global__ void quantize_x_kernel(const float* __restrict__ x,
                                  const float* __restrict__ scale_p,
                                  const int* __restrict__ zp_p,
                                  int* __restrict__ xq4, long n4)
{
    long i = (long)blockIdx.x * blockDim.x + threadIdx.x;
    if (i >= n4) return;
    const float s  = *scale_p;
    const float zp = (float)(*zp_p);
    float4 v = ((const float4*)x)[i];
    float q0 = fminf(fmaxf(rintf(v.x / s) + zp, -128.0f), 127.0f);
    float q1 = fminf(fmaxf(rintf(v.y / s) + zp, -128.0f), 127.0f);
    float q2 = fminf(fmaxf(rintf(v.z / s) + zp, -128.0f), 127.0f);
    float q3 = fminf(fmaxf(rintf(v.w / s) + zp, -128.0f), 127.0f);
    int p = ((int)q0 & 255) | (((int)q1 & 255) << 8) |
            (((int)q2 & 255) << 16) | ((int)q3 << 24);
    xq4[i] = p;
}

__global__ void pack_w_kernel(const int* __restrict__ w,
                              int* __restrict__ wq4, int n4)
{
    int i = blockIdx.x * blockDim.x + threadIdx.x;
    if (i >= n4) return;
    int4 v = ((const int4*)w)[i];
    wq4[i] = (v.x & 255) | ((v.y & 255) << 8) |
             ((v.z & 255) << 16) | ((v.w & 255) << 24);
}

__device__ __forceinline__ void stage_tile(
    const signed char* __restrict__ xq, const signed char* __restrict__ wq,
    signed char* As, signed char* Bs,
    int m0, int n0, int k0, int K, int tid)
{
#if USE_ASYNC_LDS
    #pragma unroll
    for (int it = 0; it < 2; ++it) {
        int idx = tid + it * 256;
        int row = idx >> 2;
        int col = (idx & 3) << 4;
        __builtin_amdgcn_global_load_async_to_lds_b128(
            (gas_v4i_ptr)(xq + (size_t)(m0 + row) * K + k0 + col),
            (las_v4i_ptr)(As + row * LDA_S + col),
            0, 0);
        __builtin_amdgcn_global_load_async_to_lds_b128(
            (gas_v4i_ptr)(wq + (size_t)(n0 + row) * K + k0 + col),
            (las_v4i_ptr)(Bs + row * LDB_S + col),
            0, 0);
    }
#else
    int4 ra[2], rb[2];
    #pragma unroll
    for (int it = 0; it < 2; ++it) {
        int idx = tid + it * 256;
        int row = idx >> 2;
        int col = (idx & 3) << 4;
        ra[it] = *(const int4*)(xq + (size_t)(m0 + row) * K + k0 + col);
        rb[it] = *(const int4*)(wq + (size_t)(n0 + row) * K + k0 + col);
    }
    #pragma unroll
    for (int it = 0; it < 2; ++it) {
        int idx = tid + it * 256;
        int row = idx >> 2;
        int col = (idx & 3) << 4;
        *(int4*)(As + row * LDA_S + col) = ra[it];
        *(int4*)(Bs + row * LDB_S + col) = rb[it];
    }
#endif
}

__global__ __launch_bounds__(256)
void qgemm_wmma_iu8_kernel(const signed char* __restrict__ xq,
                           const signed char* __restrict__ wq,
                           const int* __restrict__ bias,
                           const float* __restrict__ out_scale_p,
                           const int* __restrict__ out_zp_p,
                           float* __restrict__ out,
                           int M, int N, int K)
{
    __shared__ signed char As[2][BM * LDA_S];
    __shared__ signed char Bs[2][BN * LDB_S];

    const int tid   = threadIdx.x;
    const int lane  = tid & 31;
    const int wave  = tid >> 5;
    const int wm    = wave & 3;
    const int wn    = wave >> 2;
    const int l16   = lane & 15;
    const int lhalf = lane >> 4;

    const int m0 = blockIdx.y * BM;
    const int n0 = blockIdx.x * BN;

    v8i acc[2][4];
    const v8i vzero = {};
    #pragma unroll
    for (int i = 0; i < 2; ++i)
        #pragma unroll
        for (int j = 0; j < 4; ++j)
            acc[i][j] = vzero;

    stage_tile(xq, wq, As[0], Bs[0], m0, n0, 0, K, tid);

    const int T = K / BK;
    for (int t = 0; t < T; ++t) {
#if USE_ASYNC_LDS
        __builtin_amdgcn_s_wait_asynccnt(0);
#endif
        __syncthreads();
        if (t + 1 < T)
            stage_tile(xq, wq, As[(t + 1) & 1], Bs[(t + 1) & 1],
                       m0, n0, (t + 1) * BK, K, tid);

        const signed char* as = As[t & 1];
        const signed char* bs = Bs[t & 1];

        union Frag { long long d[4]; int4 q[2]; v8i v; };
        Frag afr[2], bfr[4];

        #pragma unroll
        for (int mi = 0; mi < 2; ++mi) {
            const signed char* ap =
                &as[(wm * 32 + mi * 16 + l16) * LDA_S + lhalf * 8];
            afr[mi].d[0] = *(const long long*)(ap +  0);
            afr[mi].d[1] = *(const long long*)(ap + 16);
            afr[mi].d[2] = *(const long long*)(ap + 32);
            afr[mi].d[3] = *(const long long*)(ap + 48);
        }
        #pragma unroll
        for (int ni = 0; ni < 4; ++ni) {
            const signed char* bp =
                &bs[(wn * 64 + ni * 16 + l16) * LDB_S + lhalf * 8];
            bfr[ni].d[0] = *(const long long*)(bp +  0);
            bfr[ni].d[1] = *(const long long*)(bp + 16);
            bfr[ni].d[2] = *(const long long*)(bp + 32);
            bfr[ni].d[3] = *(const long long*)(bp + 48);
        }

        #pragma unroll
        for (int mi = 0; mi < 2; ++mi)
            #pragma unroll
            for (int ni = 0; ni < 4; ++ni)
                acc[mi][ni] = __builtin_amdgcn_wmma_i32_16x16x64_iu8(
                     true, afr[mi].v,
                     true, bfr[ni].v,
                    acc[mi][ni],  false,  false);
    }

    const float oscale = *out_scale_p;
    const int   ozp    = *out_zp_p;
    #pragma unroll
    for (int ni = 0; ni < 4; ++ni) {
        const int n_col = n0 + wn * 64 + ni * 16 + l16;
        const int bn    = bias[n_col] - ozp;
        #pragma unroll
        for (int mi = 0; mi < 2; ++mi) {
            #pragma unroll
            for (int r = 0; r < 8; ++r) {
                const int m_row = m0 + wm * 32 + mi * 16 + lhalf * 8 + r;
                out[(size_t)m_row * N + n_col] =
                    (float)(acc[mi][ni][r] + bn) * oscale;
            }
        }
    }
}

extern "C" void kernel_launch(void* const* d_in, const int* in_sizes, int n_in,
                              void* d_out, int out_size, void* d_ws, size_t ws_size,
                              hipStream_t stream)
{
    const float* x         = (const float*)d_in[0];
    const int*   w         = (const int*)  d_in[1];
    const int*   bias      = (const int*)  d_in[2];
    const float* in_scale  = (const float*)d_in[3];
    const float* out_scale = (const float*)d_in[4];
    const int*   in_zp     = (const int*)  d_in[5];
    const int*   out_zp    = (const int*)  d_in[6];

    const int  N  = in_sizes[2];
    const int  K  = in_sizes[1] / N;
    const long MK = in_sizes[0];
    const int  M  = (int)(MK / K);

    signed char* xq = (signed char*)d_ws;
    signed char* wq = xq + (size_t)M * K;

    const long n4x = MK / 4;
    quantize_x_kernel<<<(unsigned)((n4x + 255) / 256), 256, 0, stream>>>(
        x, in_scale, in_zp, (int*)xq, n4x);

    const int n4w = in_sizes[1] / 4;
    pack_w_kernel<<<(n4w + 255) / 256, 256, 0, stream>>>(w, (int*)wq, n4w);

    dim3 grid(N / BN, M / BM);
    qgemm_wmma_iu8_kernel<<<grid, 256, 0, stream>>>(
        xq, wq, bias, out_scale, out_zp, (float*)d_out, M, N, K);
}
